// ParallelMixBlock_47150150975771
// MI455X (gfx1250) — matrix operand order repaired and checked by reading, not compiled, not run on hardware
//
#include <hip/hip_runtime.h>
#include <hip/hip_bf16.h>

typedef _Float16 v16h __attribute__((ext_vector_type(16)));
typedef _Float16 v8h  __attribute__((ext_vector_type(8)));
typedef float    v8f  __attribute__((ext_vector_type(8)));

#define R_TOK 200704            // 64 * 56 * 56 tokens (= 4096 windows * 49)
#define LN_EPS 1e-5f

// ---------------------------------------------------------------- helpers
__device__ __forceinline__ float gelu_f(float x) {
  return 0.5f * x * (1.0f + tanhf(0.7978845608028654f * (x + 0.044715f * x * x * x)));
}
__device__ __forceinline__ float sigmoid_f(float x) { return 1.0f / (1.0f + expf(-x)); }

__device__ __forceinline__ float block_reduce_sum(float v, float* sm) {
  int tid = threadIdx.x, n = blockDim.x;
  sm[tid] = v; __syncthreads();
  for (int s = n >> 1; s > 0; s >>= 1) {
    if (tid < s) sm[tid] += sm[tid + s];
    __syncthreads();
  }
  float r = sm[0]; __syncthreads();
  return r;
}

// token (b, y, x) order -> window-major row  r = ((b*8+wh)*8+ww)*49 + i*7 + j
__device__ __forceinline__ long tok2win(int t) {
  int b = t / 3136, pix = t % 3136;
  int y = pix / 56, x = pix % 56;
  return (long)((b * 8 + y / 7) * 8 + x / 7) * 49 + (y % 7) * 7 + (x % 7);
}

// ---------------------------------------------------------------- weight swizzle
// Pack f32 weight (K x N, optionally stored transposed as N x K) into the per-lane
// B-fragment order used here for v_wmma_f32_16x16x32_f16 (contiguous k, same as the A loads):
//   flat = ((ntile*Kt + ktile)*32 + lane)*16 + e
//   lane 0-15  : K = ktile*32 + e        , N = ntile*16 + lane
//   lane 16-31 : K = ktile*32 + 16 + e   , N = ntile*16 + (lane-16)
__global__ void cvt_frag_kernel(const float* __restrict__ w, _Float16* __restrict__ out,
                                int K, int N, int trans) {
  int f = blockIdx.x * 256 + threadIdx.x;
  if (f >= K * N) return;
  int Kt = K >> 5;
  int e = f & 15;
  int lane = (f >> 4) & 31;
  int kt = (f >> 9) % Kt;
  int nt = f / (512 * Kt);
  int k = kt * 32 + (lane >> 4) * 16 + e;
  int n = nt * 16 + (lane & 15);
  float v = trans ? w[(size_t)n * K + k] : w[(size_t)k * N + n];
  out[f] = (_Float16)v;
}

// ---------------------------------------------------------------- WMMA GEMM
// C(M x N) = A(M x K, f16 row-major) @ Wfrag(K x N, pre-swizzled f16) + bias
// Block = 4 waves; wave w computes rows [blockM + 16w, +16), cols [blockN, +128).
// All 16 B-fragment b128 loads + 2 A loads are issued before the 8 WMMAs of a
// k-step; a sched_barrier keeps the scheduler from interleaving them back, so
// the loadcnt waits stagger and the next k-step's loads overlap the XDL pipe.
#define EPI_GELU 1
#define EPI_RES  2
#define EPI_F16  4

template <int EPI>
__global__ __launch_bounds__(128) void gemm_wmma_kernel(
    const _Float16* __restrict__ A, const _Float16* __restrict__ Wf,
    const float* __restrict__ bias, float* __restrict__ outF,
    _Float16* __restrict__ outH, const float* __restrict__ resid,
    int N, int K) {
  int w = threadIdx.x >> 5;
  int lane = threadIdx.x & 31;
  int half_ = lane >> 4;
  int lr = lane & 15;
  int rowBase = blockIdx.x * 64 + w * 16;
  int colBase = blockIdx.y * 128;
  int Kt = K >> 5;

  v8f acc[8] = {};
  const _Float16* aRow = A + (size_t)(rowBase + lr) * K;
  // per-lane base into the swizzled weight fragments for this block's 8 n-tiles
  const _Float16* wBase = Wf + ((size_t)((colBase >> 4) * Kt) * 32 + lane) * 16;

  for (int k0 = 0; k0 < K; k0 += 32) {
    int kt = k0 >> 5;
    // issue the whole batch of loads for this k-step first
    v8h a0 = *(const v8h*)(aRow + k0 + half_ * 16);
    v8h a1 = *(const v8h*)(aRow + k0 + half_ * 16 + 8);
    v16h b[8];
#pragma unroll
    for (int nt = 0; nt < 8; nt++) {
      b[nt] = *(const v16h*)(wBase + ((size_t)(nt * Kt + kt) * 32) * 16);
    }
    v16h a = __builtin_shufflevector(a0, a1, 0, 1, 2, 3, 4, 5, 6, 7,
                                     8, 9, 10, 11, 12, 13, 14, 15);
    // keep all loads above, all WMMAs below
    __builtin_amdgcn_sched_barrier(0);
#pragma unroll
    for (int nt = 0; nt < 8; nt++) {
      acc[nt] = __builtin_amdgcn_wmma_f32_16x16x32_f16(
          false, a, false, b[nt], (short)0, acc[nt], false, false);
    }
  }

  // C/D layout: VGPR i, lanes 0-15: M=i, N=lane ; lanes 16-31: M=i+8, N=lane-16
#pragma unroll
  for (int nt = 0; nt < 8; nt++) {
    int col = colBase + nt * 16 + lr;
    float bv = bias[col];
#pragma unroll
    for (int i = 0; i < 8; i++) {
      int row = rowBase + half_ * 8 + i;
      float v = acc[nt][i] + bv;
      if (EPI & EPI_GELU) v = gelu_f(v);
      if (EPI & EPI_RES) v += resid[(size_t)row * N + col];
      if (EPI & EPI_F16)
        outH[(size_t)row * N + col] = (_Float16)v;
      else
        outF[(size_t)row * N + col] = v;
    }
  }
}

// ---------------------------------------------------------------- LN kernels
// LN1: x (token order, f32) -> hw16 (window order, f16)
__global__ void ln1_kernel(const float* __restrict__ x, const float* __restrict__ g,
                           const float* __restrict__ b, _Float16* __restrict__ hw) {
  __shared__ float sm[256];
  int t = blockIdx.x, c = threadIdx.x;
  float v = x[(size_t)t * 256 + c];
  float mean = block_reduce_sum(v, sm) * (1.0f / 256.0f);
  float d = v - mean;
  float var = block_reduce_sum(d * d, sm) * (1.0f / 256.0f);
  float o = d * rsqrtf(var + LN_EPS) * g[c] + b[c];
  long r = tok2win(t);
  hw[r * 256 + c] = (_Float16)o;
}

// generic row LN: in f32 (R x C) -> out (f32 or f16) with stride / column offset
__global__ void ln_rows_kernel(const float* __restrict__ in, const float* __restrict__ g,
                               const float* __restrict__ b, float* outF,
                               _Float16* outH, int C, int outStride, int outOff) {
  extern __shared__ float sm[];
  long r = blockIdx.x;
  int c = threadIdx.x;
  float v = in[r * C + c];
  float inv = 1.0f / (float)C;
  float mean = block_reduce_sum(v, sm) * inv;
  float d = v - mean;
  float var = block_reduce_sum(d * d, sm) * inv;
  float o = d * rsqrtf(var + LN_EPS) * g[c] + b[c];
  size_t oi = r * (size_t)outStride + outOff + c;
  if (outH) outH[oi] = (_Float16)o;
  else outF[oi] = o;
}

// unwindow + residual add + LN2: x2 = x + aw(perm); y16 = LN2(x2)
__global__ void add_ln2_kernel(const float* __restrict__ x, const float* __restrict__ aw,
                               const float* __restrict__ g, const float* __restrict__ b,
                               float* __restrict__ x2, _Float16* __restrict__ y16) {
  __shared__ float sm[256];
  int t = blockIdx.x, c = threadIdx.x;
  long r = tok2win(t);
  float v = x[(size_t)t * 256 + c] + aw[r * 256 + c];
  x2[(size_t)t * 256 + c] = v;
  float mean = block_reduce_sum(v, sm) * (1.0f / 256.0f);
  float d = v - mean;
  float var = block_reduce_sum(d * d, sm) * (1.0f / 256.0f);
  y16[(size_t)t * 256 + c] = (_Float16)(d * rsqrtf(var + LN_EPS) * g[c] + b[c]);
}

// ---------------------------------------------------------------- conv branch
// depthwise 3x3 + BN + gelu, all in window-row layout, zero-padded borders
__global__ void dwconv_kernel(const float* __restrict__ in, const float* __restrict__ w,
                              const float* __restrict__ cb, const float* __restrict__ bg,
                              const float* __restrict__ bb, const float* __restrict__ bm,
                              const float* __restrict__ bv, _Float16* __restrict__ out) {
  long idx = (long)blockIdx.x * 256 + threadIdx.x;  // r*256 + c
  int c = (int)(idx & 255);
  long r = idx >> 8;
  int win = (int)(r / 49), n = (int)(r % 49);
  int b = win >> 6, wh = (win >> 3) & 7, ww = win & 7;
  int y = wh * 7 + n / 7, x = ww * 7 + n % 7;
  float acc = 0.0f;
#pragma unroll
  for (int dy = -1; dy <= 1; dy++)
#pragma unroll
    for (int dx = -1; dx <= 1; dx++) {
      int yy = y + dy, xx = x + dx;
      if (yy >= 0 && yy < 56 && xx >= 0 && xx < 56) {
        long r2 = (long)((b * 8 + yy / 7) * 8 + xx / 7) * 49 + (yy % 7) * 7 + (xx % 7);
        acc += w[c * 9 + (dy + 1) * 3 + (dx + 1)] * in[r2 * 256 + c];
      }
    }
  acc += cb[c];
  acc = (acc - bm[c]) * rsqrtf(bv[c] + LN_EPS) * bg[c] + bb[c];
  out[idx] = (_Float16)gelu_f(acc);
}

// adaptive average pool over the 3136 tokens of each batch (rows contiguous per batch)
__global__ void ci_pool_kernel(const _Float16* __restrict__ dw, float* __restrict__ ci0) {
  int b = blockIdx.x, c = threadIdx.x;  // 256 threads
  size_t base = (size_t)b * 3136 * 256 + c;
  float s = 0.0f;
  for (int i = 0; i < 3136; i++) s += (float)dw[base + (size_t)i * 256];
  ci0[b * 256 + c] = s * (1.0f / 3136.0f);
}

// channel interaction MLP: 256 -> 32 (BN+gelu) -> 128 ; 64 batches, tiny
__global__ __launch_bounds__(128) void ci_mlp_kernel(
    const float* __restrict__ ci0, const float* __restrict__ w1, const float* __restrict__ b1,
    const float* __restrict__ g, const float* __restrict__ bb, const float* __restrict__ m,
    const float* __restrict__ vv, const float* __restrict__ w2, const float* __restrict__ b2,
    float* __restrict__ ci) {
  __shared__ float row[256], t[32];
  int b = blockIdx.x, tid = threadIdx.x;
  for (int i = tid; i < 256; i += 128) row[i] = ci0[b * 256 + i];
  __syncthreads();
  if (tid < 32) {
    float a = 0.0f;
    for (int k = 0; k < 256; k++) a += w1[tid * 256 + k] * row[k];  // ci1_w (32,256)
    a += b1[tid];
    a = (a - m[tid]) * rsqrtf(vv[tid] + LN_EPS) * g[tid] + bb[tid];
    t[tid] = gelu_f(a);
  }
  __syncthreads();
  if (tid < 128) {
    float a = 0.0f;
    for (int j = 0; j < 32; j++) a += w2[tid * 32 + j] * t[j];      // ci2_w (128,32)
    ci[b * 128 + tid] = a + b2[tid];
  }
}

// ---------------------------------------------------------------- attention
// one block per (window, head); N=49, HD=16; gate on V, rel-pos bias, softmax
__global__ __launch_bounds__(64) void attn_kernel(
    const float* __restrict__ qkv, const float* __restrict__ ci,
    const float* __restrict__ rpb, float* __restrict__ out) {
  __shared__ float q[49][16], k[49][16], v[49][16], s[49][49];
  int win = blockIdx.x, h = blockIdx.y;
  int bidx = win >> 6;
  int tid = threadIdx.x;
  const float scale = 0.25f;  // 16^-0.5
  for (int p = tid; p < 49 * 16; p += 64) {
    int n = p >> 4, d = p & 15;
    size_t ro = (size_t)(win * 49 + n) * 384;
    q[n][d] = qkv[ro + h * 16 + d] * scale;
    k[n][d] = qkv[ro + 128 + h * 16 + d];
    float gate = sigmoid_f(ci[bidx * 128 + h * 16 + d]);
    v[n][d] = qkv[ro + 256 + h * 16 + d] * gate;
  }
  __syncthreads();
  for (int p = tid; p < 49 * 49; p += 64) {
    int n = p / 49, m = p % 49;
    float a = 0.0f;
#pragma unroll
    for (int d = 0; d < 16; d++) a += q[n][d] * k[m][d];
    int dh = n / 7 - m / 7 + 6, dw = n % 7 - m % 7 + 6;
    s[n][m] = a + rpb[(dh * 13 + dw) * 8 + h];
  }
  __syncthreads();
  if (tid < 49) {
    float mx = -1e30f;
    for (int m = 0; m < 49; m++) mx = fmaxf(mx, s[tid][m]);
    float sum = 0.0f;
    for (int m = 0; m < 49; m++) { float e = expf(s[tid][m] - mx); s[tid][m] = e; sum += e; }
    float inv = 1.0f / sum;
    for (int m = 0; m < 49; m++) s[tid][m] *= inv;
  }
  __syncthreads();
  for (int p = tid; p < 49 * 16; p += 64) {
    int n = p >> 4, d = p & 15;
    float a = 0.0f;
    for (int m = 0; m < 49; m++) a += s[n][m] * v[m][d];
    out[(size_t)(win * 49 + n) * 128 + h * 16 + d] = a;
  }
}

// ---------------------------------------------------------------- spatial interaction
// si1: xatt (R x 128) @ si1_w^T (128 -> 16) + bias, BN, gelu
__global__ void si1_kernel(const float* __restrict__ xatt, const float* __restrict__ w,
                           const float* __restrict__ b, const float* __restrict__ g,
                           const float* __restrict__ bb, const float* __restrict__ m,
                           const float* __restrict__ vv, float* __restrict__ out) {
  long idx = (long)blockIdx.x * 256 + threadIdx.x;  // r*16 + c
  int c = (int)(idx & 15);
  long r = idx >> 4;
  const float* xr = xatt + r * 128;
  float a = 0.0f;
  for (int k = 0; k < 128; k++) a += w[c * 128 + k] * xr[k];  // si1_w (16,128)
  a += b[c];
  a = (a - m[c]) * rsqrtf(vv[c] + LN_EPS) * g[c] + bb[c];
  out[idx] = gelu_f(a);
}

// si2 + sigmoid gate + conv_bn, write left half of concat buffer (f16)
__global__ __launch_bounds__(128) void gate_bn_kernel(
    const float* __restrict__ prj, const float* __restrict__ si1,
    const float* __restrict__ w2, const float* __restrict__ b2,
    const float* __restrict__ g, const float* __restrict__ bb,
    const float* __restrict__ m, const float* __restrict__ vv,
    _Float16* __restrict__ cat) {
  long r = blockIdx.x;
  int c = threadIdx.x;
  float s = 0.0f;
#pragma unroll
  for (int j = 0; j < 16; j++) s += w2[j] * si1[r * 16 + j];  // si2_w (1,16)
  s = sigmoid_f(s + b2[0]);
  float v = prj[r * 128 + c] * s;
  v = (v - m[c]) * rsqrtf(vv[c] + LN_EPS) * g[c] + bb[c];
  cat[r * 256 + c] = (_Float16)v;
}

// ---------------------------------------------------------------- host
enum {
  IN_X = 0, IN_N1G, IN_N1B,
  IN_PAW, IN_PAB, IN_PALNG, IN_PALNB,
  IN_PCW, IN_PCB, IN_PCLNG, IN_PCLNB,
  IN_DWW, IN_DWB, IN_DWBNG, IN_DWBNB, IN_DWBNM, IN_DWBNV,
  IN_CI1W, IN_CI1B, IN_CIBNG, IN_CIBNB, IN_CIBNM, IN_CIBNV,
  IN_CI2W, IN_CI2B,
  IN_PRJW, IN_PRJB,
  IN_QKVW, IN_QKVB,
  IN_SI1W, IN_SI1B, IN_SIBNG, IN_SIBNB, IN_SIBNM, IN_SIBNV,
  IN_SI2W, IN_SI2B,
  IN_CBNG, IN_CBNB, IN_CBNM, IN_CBNV,
  IN_ALNG, IN_ALNB,
  IN_PROJW, IN_PROJB,
  IN_RPB,
  IN_N2G, IN_N2B,
  IN_FC1W, IN_FC1B, IN_FC2W, IN_FC2B,
  IN_HH, IN_WW
};

extern "C" void kernel_launch(void* const* d_in, const int* in_sizes, int n_in,
                              void* d_out, int out_size, void* d_ws, size_t ws_size,
                              hipStream_t stream) {
  (void)in_sizes; (void)n_in; (void)out_size; (void)ws_size;
  const long R = R_TOK;
  auto F = [&](int i) { return (const float*)d_in[i]; };
  float* out = (float*)d_out;

  char* ws = (char*)d_ws;
  size_t off = 0;
  auto alloc = [&](size_t bytes) {
    size_t o = off;
    off = (off + bytes + 255) & ~(size_t)255;
    return o;
  };

  size_t o_hw16  = alloc(R * 256 * 2);          // LN1 output, window order (later y16)
  size_t o_buf1  = alloc(R * 256 * 4);          // pc_out / xcnn_ln / later aw (proj out)
  size_t o_scr   = alloc(R * 256 * 4 +          // region aliased by fc1 output
                         R * 128 * 2 + R * 256 * 2 + R * 128 * 4 + R * 128 * 4);
  size_t o_pa    = o_scr;                       // pa_out f32      R*128*4
  size_t o_xa16  = o_pa + R * 128 * 4;          // x_atten f16     R*128*2
  size_t o_dw16  = o_xa16 + R * 128 * 2;        // dwconv out f16  R*256*2
  size_t o_prj   = o_dw16 + R * 256 * 2;        // prj out f32     R*128*4
  size_t o_xatt2 = o_prj + R * 128 * 4;         // attention out   R*128*4
  size_t o_h1    = o_scr;                       // fc1 f16 out aliases the whole region
  size_t o_qkv   = alloc(R * 384 * 4);          // qkv f32 (later x2 f32)
  size_t o_x2    = o_qkv;
  size_t o_si1   = alloc(R * 16 * 4);
  size_t o_cat16 = alloc(R * 256 * 2);
  size_t o_ci0   = alloc(64 * 256 * 4);
  size_t o_ci    = alloc(64 * 128 * 4);
  // pre-swizzled f16 weight fragments
  size_t o_wpa   = alloc(256 * 128 * 2);
  size_t o_wpc   = alloc(256 * 256 * 2);
  size_t o_wqkv  = alloc(128 * 384 * 2);
  size_t o_wprj  = alloc(256 * 128 * 2);
  size_t o_wproj = alloc(256 * 256 * 2);
  size_t o_wfc1  = alloc(256 * 1024 * 2);
  size_t o_wfc2  = alloc(1024 * 256 * 2);

  auto H16 = [&](size_t o) { return (_Float16*)(ws + o); };
  auto F32 = [&](size_t o) { return (float*)(ws + o); };

  // --- weight conversion (f32 -> swizzled WMMA B fragments, f16)
  auto cvt = [&](int idx, size_t o, int K, int N, int trans) {
    int tot = K * N;
    cvt_frag_kernel<<<(tot + 255) / 256, 256, 0, stream>>>(F(idx), H16(o), K, N, trans);
  };
  cvt(IN_PAW, o_wpa, 256, 128, 0);
  cvt(IN_PCW, o_wpc, 256, 256, 0);
  cvt(IN_QKVW, o_wqkv, 128, 384, 0);
  cvt(IN_PRJW, o_wprj, 256, 128, 1);   // stored (128,256) = (N,K)
  cvt(IN_PROJW, o_wproj, 256, 256, 0);
  cvt(IN_FC1W, o_wfc1, 256, 1024, 0);
  cvt(IN_FC2W, o_wfc2, 1024, 256, 0);

  auto gemm = [&](size_t oA, size_t oW, const float* bias, float* oF, _Float16* oH,
                  const float* res, int N, int K, int epi) {
    dim3 grid(R_TOK / 64, N / 128);
    switch (epi) {
      case 0:
        gemm_wmma_kernel<0><<<grid, 128, 0, stream>>>(H16(oA), H16(oW), bias, oF, oH, res, N, K);
        break;
      case EPI_GELU | EPI_F16:
        gemm_wmma_kernel<EPI_GELU | EPI_F16><<<grid, 128, 0, stream>>>(H16(oA), H16(oW), bias, oF, oH, res, N, K);
        break;
      case EPI_RES:
        gemm_wmma_kernel<EPI_RES><<<grid, 128, 0, stream>>>(H16(oA), H16(oW), bias, oF, oH, res, N, K);
        break;
    }
  };

  // 1. LN1 + window partition -> hw16
  ln1_kernel<<<R_TOK, 256, 0, stream>>>(F(IN_X), F(IN_N1G), F(IN_N1B), H16(o_hw16));

  // 2. pa GEMM + LN -> x_atten16
  gemm(o_hw16, o_wpa, F(IN_PAB), F32(o_pa), nullptr, nullptr, 128, 256, 0);
  ln_rows_kernel<<<R_TOK, 128, 128 * 4, stream>>>(F32(o_pa), F(IN_PALNG), F(IN_PALNB),
                                                  nullptr, H16(o_xa16), 128, 128, 0);

  // 3. pc GEMM + LN (in place, f32) -> xcnn_ln
  gemm(o_hw16, o_wpc, F(IN_PCB), F32(o_buf1), nullptr, nullptr, 256, 256, 0);
  ln_rows_kernel<<<R_TOK, 256, 256 * 4, stream>>>(F32(o_buf1), F(IN_PCLNG), F(IN_PCLNB),
                                                  F32(o_buf1), nullptr, 256, 256, 0);

  // 4. depthwise conv + BN + gelu -> dw16
  dwconv_kernel<<<R_TOK, 256, 0, stream>>>(F32(o_buf1), F(IN_DWW), F(IN_DWB),
                                           F(IN_DWBNG), F(IN_DWBNB), F(IN_DWBNM), F(IN_DWBNV),
                                           H16(o_dw16));

  // 5. channel interaction (avg pool + tiny MLP)
  ci_pool_kernel<<<64, 256, 0, stream>>>(H16(o_dw16), F32(o_ci0));
  ci_mlp_kernel<<<64, 128, 0, stream>>>(F32(o_ci0), F(IN_CI1W), F(IN_CI1B),
                                        F(IN_CIBNG), F(IN_CIBNB), F(IN_CIBNM), F(IN_CIBNV),
                                        F(IN_CI2W), F(IN_CI2B), F32(o_ci));

  // 6. projection conv 256->128 on conv branch
  gemm(o_dw16, o_wprj, F(IN_PRJB), F32(o_prj), nullptr, nullptr, 128, 256, 0);

  // 7. qkv GEMM
  gemm(o_xa16, o_wqkv, F(IN_QKVB), F32(o_qkv), nullptr, nullptr, 384, 128, 0);

  // 8. windowed attention (gate on V, rel-pos bias, softmax)
  attn_kernel<<<dim3(4096, 8), 64, 0, stream>>>(F32(o_qkv), F32(o_ci), F(IN_RPB), F32(o_xatt2));

  // 9. spatial interaction -> gate conv branch -> conv_bn -> cat[:,0:128]
  si1_kernel<<<(int)(R * 16 / 256), 256, 0, stream>>>(F32(o_xatt2), F(IN_SI1W), F(IN_SI1B),
                                                      F(IN_SIBNG), F(IN_SIBNB), F(IN_SIBNM),
                                                      F(IN_SIBNV), F32(o_si1));
  gate_bn_kernel<<<R_TOK, 128, 0, stream>>>(F32(o_prj), F32(o_si1), F(IN_SI2W), F(IN_SI2B),
                                            F(IN_CBNG), F(IN_CBNB), F(IN_CBNM), F(IN_CBNV),
                                            H16(o_cat16));

  // 10. attn LN -> cat[:,128:256]
  ln_rows_kernel<<<R_TOK, 128, 128 * 4, stream>>>(F32(o_xatt2), F(IN_ALNG), F(IN_ALNB),
                                                  nullptr, H16(o_cat16), 128, 256, 128);

  // 11. mixer projection -> aw (window order, reuses buf1)
  gemm(o_cat16, o_wproj, F(IN_PROJB), F32(o_buf1), nullptr, nullptr, 256, 256, 0);

  // 12. un-window + residual + LN2 -> x2 (token order), y16
  add_ln2_kernel<<<R_TOK, 256, 0, stream>>>(F(IN_X), F32(o_buf1), F(IN_N2G), F(IN_N2B),
                                            F32(o_x2), H16(o_hw16));

  // 13. MLP: fc1 (gelu, f16 out) then fc2 (+ residual) -> d_out
  gemm(o_hw16, o_wfc1, F(IN_FC1B), nullptr, H16(o_h1), nullptr, 1024, 256, EPI_GELU | EPI_F16);
  gemm(o_h1, o_wfc2, F(IN_FC2B), out, nullptr, F32(o_x2), 256, 1024, EPI_RES);
}
